// Model_77120432767628
// MI455X (gfx1250) — matrix operand order repaired and checked by reading, not compiled, not run on hardware
//
#include <hip/hip_runtime.h>
#include <hip/hip_bf16.h>
#include <math.h>

#define TWO_PI 6.28318530717958647692f

typedef __attribute__((ext_vector_type(16))) _Float16 v16h;
typedef __attribute__((ext_vector_type(8)))  _Float16 v8h;
typedef __attribute__((ext_vector_type(8)))  float    v8f;

enum { GF_BIAS = 1, GF_RESID = 2, GF_GELU = 4, GF_ADDC = 8 };

// =====================================================================
// Batched strided GEMM on WMMA f16->f32:  C = epilogue(A(MxK) @ B(KxN))
// FAST-PATH REQUIREMENTS (true for every call in this model):
//   sAk == 1 (A rows contiguous in K), sBn == 1 (B rows contiguous in N),
//   K % 32 == 0, N % 16 == 0, M % 128 == 0 (all waves active).
// One wave computes one 16x16 tile; 8 waves per block share one B tile
// staged through LDS (f32 -> f16 on the way in).
// =====================================================================
__global__ __launch_bounds__(256)
void k_gemm_wmma(const float* __restrict__ A, const float* __restrict__ B,
                 float* __restrict__ C, const float* __restrict__ bias,
                 const float* __restrict__ resid,
                 int M, int N, int K,
                 long long sAm, long long bsA,
                 long long sBk, long long bsB,
                 long long sCm, long long bsC,
                 int flags, float cScale)
{
    __shared__ _Float16 btile[512];            // 32(K) x 16(N), column-major

    const int lane = threadIdx.x;              // 0..31 (wave32)
    const int wave = threadIdx.y;              // 0..7
    const int tid  = wave * 32 + lane;
    const int l16  = lane & 15;
    const int hi   = lane >> 4;                // lane half
    const int tileN = blockIdx.x;
    const int tileM = blockIdx.y * 8 + wave;
    const bool active = (tileM * 16 < M);
    const long long b = blockIdx.z;

    const float* Ab = A + b * bsA;
    const float* Bb = B + b * bsB;
    float*       Cb = C + b * bsC;
    const float* Rb = resid ? (resid + b * bsC) : nullptr;

    const long long rowA = (long long)tileM * 16;
    const long long colB = (long long)tileN * 16;
    const float* aRow = Ab + (rowA + l16) * sAm;

    // B staging coordinates: each thread loads 2 consecutive floats of a row
    const int idx2 = tid * 2;
    const int brow = idx2 >> 4;                // 0..31  (K within tile)
    const int bcol = idx2 & 15;                // 0,2,..,14

    v8f acc = {0.f, 0.f, 0.f, 0.f, 0.f, 0.f, 0.f, 0.f};

    for (int k0 = 0; k0 < K; k0 += 32) {
        // ---- cooperative, coalesced B-tile load -> f16 LDS (col-major) ----
        {
            const float* bsrc = Bb + (long long)(k0 + brow) * sBk + colB + bcol;
            float2 bv = *(const float2*)bsrc;
            btile[bcol * 32 + brow]       = (_Float16)bv.x;
            btile[(bcol + 1) * 32 + brow] = (_Float16)bv.y;
        }
        __syncthreads();

        if (active) {
            // ---- A fragment: 4x float4 loads (contiguous K) ----
            v16h af;
            {
                const float4* a0p = (const float4*)(aRow + k0 + hi * 8);
                const float4* a1p = (const float4*)(aRow + k0 + 16 + hi * 8);
                float4 q0 = a0p[0], q1 = a0p[1];
                float4 q2 = a1p[0], q3 = a1p[1];
                af[0]  = (_Float16)q0.x; af[1]  = (_Float16)q0.y;
                af[2]  = (_Float16)q0.z; af[3]  = (_Float16)q0.w;
                af[4]  = (_Float16)q1.x; af[5]  = (_Float16)q1.y;
                af[6]  = (_Float16)q1.z; af[7]  = (_Float16)q1.w;
                af[8]  = (_Float16)q2.x; af[9]  = (_Float16)q2.y;
                af[10] = (_Float16)q2.z; af[11] = (_Float16)q2.w;
                af[12] = (_Float16)q3.x; af[13] = (_Float16)q3.y;
                af[14] = (_Float16)q3.z; af[15] = (_Float16)q3.w;
            }
            // ---- B fragment: two b128 DS reads from the staged column ----
            v16h bf;
            {
                const v8h* bp = (const v8h*)&btile[l16 * 32 + hi * 8];
                v8h b0 = bp[0], b1 = bp[2];
                #pragma unroll
                for (int i = 0; i < 8; ++i) { bf[i] = b0[i]; bf[i + 8] = b1[i]; }
            }
            acc = __builtin_amdgcn_wmma_f32_16x16x32_f16(false, af, false, bf,
                                                         (short)0, acc, false, false);
        }
        __syncthreads();
    }

    if (!active) return;
    const long long n = colB + l16;
    #pragma unroll
    for (int r = 0; r < 8; ++r) {
        long long m = rowA + r + hi * 8;       // C layout: VGPR r -> M=r (+8 for hi)
        float v = acc[r] * cScale;
        if (flags & GF_ADDC)  v += Cb[m * sCm + n];
        if (flags & GF_BIAS)  v += bias[n];
        if (flags & GF_GELU)  v = 0.5f * v * (1.f + erff(v * 0.70710678118f));
        if (flags & GF_RESID) v += Rb[m * sCm + n];
        Cb[m * sCm + n] = v;
    }
}

static inline void gemm(hipStream_t st, const float* A, const float* B, float* C,
                        const float* bias, const float* resid,
                        int M, int N, int K,
                        long long sAm, long long bsA,
                        long long sBk, long long bsB,
                        long long sCm, long long bsC,
                        int batch, int flags, float cScale)
{
    dim3 blk(32, 8);
    dim3 grd(N / 16, (M / 16 + 7) / 8, batch);
    k_gemm_wmma<<<grd, blk, 0, st>>>(A, B, C, bias, resid, M, N, K,
                                     sAm, bsA, sBk, bsB, sCm, bsC, flags, cScale);
}

// =====================================================================
// series_decomp: moving mean (k=25, replicate pad) -> seasonal / trend
// tmode: 0 = discard trend, 1 = store trend, 2 = accumulate trend
// =====================================================================
__global__ __launch_bounds__(256)
void k_decomp(const float* __restrict__ x, float* __restrict__ seas,
              float* __restrict__ trend, int L, int C, int tmode, int total)
{
    int i = blockIdx.x * blockDim.x + threadIdx.x;
    if (i >= total) return;
    int c = i % C;
    int t = (i / C) % L;
    int b = i / (C * L);
    const float* xb = x + (long long)b * L * C + c;
    float s = 0.f;
    #pragma unroll 5
    for (int j = -12; j <= 12; ++j) {
        int tj = t + j;
        tj = tj < 0 ? 0 : (tj >= L ? L - 1 : tj);
        s += xb[(long long)tj * C];
    }
    float mean = s * (1.f / 25.f);
    if (seas) seas[i] = x[i] - mean;
    if (tmode == 1) trend[i] = mean;
    else if (tmode == 2) trend[i] += mean;
}

// embedding: circular conv1d(k=3) over CIN channels + time-mark projection
__global__ __launch_bounds__(256)
void k_embed(const float* __restrict__ xin, const float* __restrict__ wc,
             const float* __restrict__ mark, const float* __restrict__ wt,
             float* __restrict__ out, int L, int Ci, int NMk, int total)
{
    int i = blockIdx.x * blockDim.x + threadIdx.x;
    if (i >= total) return;
    int co = i & 511;
    int t  = (i >> 9) % L;
    int b  = i / (512 * L);
    const float* xb = xin + (long long)b * L * Ci;
    float acc = 0.f;
    for (int d = 0; d < 3; ++d) {
        int ts = t + d - 1;
        ts = ts < 0 ? ts + L : (ts >= L ? ts - L : ts);
        const float* xr = xb + (long long)ts * Ci;
        const float* wr = wc + (long long)(d * Ci) * 512 + co;
        for (int ci = 0; ci < Ci; ++ci) acc += xr[ci] * wr[(long long)ci * 512];
    }
    const float* mb = mark + ((long long)b * L + t) * NMk;
    for (int j = 0; j < NMk; ++j) acc += mb[j] * wt[j * 512 + co];
    out[i] = acc;
}

// trend projection: circular conv1d(k=3) 512 -> 21, accumulated into trend
__global__ __launch_bounds__(256)
void k_trend_conv(const float* __restrict__ tin, const float* __restrict__ w,
                  float* __restrict__ trend, int total)
{
    int i = blockIdx.x * blockDim.x + threadIdx.x;
    if (i >= total) return;
    int c = i % 21;
    int t = (i / 21) & 511;
    int b = i / (21 * 512);
    const float* tb = tin + (long long)b * 512 * 512;
    float acc = 0.f;
    for (int d = 0; d < 3; ++d) {
        int ts = t + d - 1;
        ts = ts < 0 ? ts + 512 : (ts >= 512 ? ts - 512 : ts);
        const float* xr = tb + (long long)ts * 512;
        const float* wr = w + (long long)d * 512 * 21 + c;
        for (int k = 0; k < 512; ++k) acc += xr[k] * wr[k * 21];
    }
    trend[i] += acc;
}

// LayerNorm over channel dim (512); one block per (b,t)
__global__ __launch_bounds__(256)
void k_layernorm(const float* __restrict__ x, float* __restrict__ y,
                 const float* __restrict__ g, const float* __restrict__ bta)
{
    __shared__ float red[256];
    long long row = blockIdx.x;
    const float* xr = x + row * 512;
    float* yr = y + row * 512;
    int tid = threadIdx.x;
    float s = xr[tid] + xr[tid + 256];
    red[tid] = s; __syncthreads();
    for (int o = 128; o > 0; o >>= 1) { if (tid < o) red[tid] += red[tid + o]; __syncthreads(); }
    float mu = red[0] * (1.f / 512.f); __syncthreads();
    float d0 = xr[tid] - mu, d1 = xr[tid + 256] - mu;
    red[tid] = d0 * d0 + d1 * d1; __syncthreads();
    for (int o = 128; o > 0; o >>= 1) { if (tid < o) red[tid] += red[tid + o]; __syncthreads(); }
    float rstd = rsqrtf(red[0] * (1.f / 512.f) + 1e-5f);
    yr[tid]       = d0 * rstd * g[tid] + bta[tid];
    yr[tid + 256] = d1 * rstd * g[tid + 256] + bta[tid + 256];
}

// my_Layernorm tail: subtract per-(b,c) mean over time
__global__ __launch_bounds__(256)
void k_sub_timemean(const float* __restrict__ x, float* __restrict__ y, int total)
{
    int i = blockIdx.x * blockDim.x + threadIdx.x;
    if (i >= total) return;
    int c = i & 511;
    int b = i >> 9;
    const float* xb = x + (long long)b * 512 * 512 + c;
    float s = 0.f;
    for (int t = 0; t < 512; ++t) s += xb[(long long)t * 512];
    float mu = s * (1.f / 512.f);
    float* yb = y + (long long)b * 512 * 512 + c;
    for (int t = 0; t < 512; ++t) yb[(long long)t * 512] = xb[(long long)t * 512] - mu;
}

// trend init: first 256 rows from trend0 tail, rest = mean over time of x_enc
__global__ void k_trend_init(const float* __restrict__ xenc, const float* __restrict__ tr0,
                             float* __restrict__ trend, int total)
{
    int i = blockIdx.x * blockDim.x + threadIdx.x;
    if (i >= total) return;
    int c = i % 21;
    int b = i / 21;
    const float* xb = xenc + (long long)b * 512 * 21 + c;
    float s = 0.f;
    for (int t = 0; t < 512; ++t) s += xb[t * 21];
    float mean = s * (1.f / 512.f);
    float* tb = trend + (long long)b * 512 * 21 + c;
    const float* t0 = tr0 + (long long)b * 512 * 21 + c;
    for (int t = 0; t < 256; ++t) tb[t * 21] = t0[(256 + t) * 21];
    for (int t = 256; t < 512; ++t) tb[t * 21] = mean;
}

// decoder seasonal input: tail of encoder seasonal, zero-padded
__global__ void k_seasonal_init(const float* __restrict__ seas, float* __restrict__ out, int total)
{
    int i = blockIdx.x * blockDim.x + threadIdx.x;
    if (i >= total) return;
    int c = i % 21;
    int t = (i / 21) & 511;
    int b = i / (21 * 512);
    out[i] = (t < 256) ? seas[((long long)b * 512 + 256 + t) * 21 + c] : 0.f;
}

// mark = concat(x_mark_enc[:, -256:], x_mark_dec)
__global__ void k_mark_concat(const float* __restrict__ me, const float* __restrict__ md,
                              float* __restrict__ out, int total)
{
    int i = blockIdx.x * blockDim.x + threadIdx.x;
    if (i >= total) return;
    int c = i & 3;
    int t = (i >> 2) & 511;
    int b = i >> 11;
    out[i] = (t < 256) ? me[(((long long)b * 512) + 256 + t) * 4 + c]
                       : md[(((long long)b * 256) + (t - 256)) * 4 + c];
}

// Unified basis builder (row-major (128,512)):
//   out[m, t]    =  s*cos(2*pi*k*t/512) * scale
//   out[64+m, t] = -s*sin(2*pi*k*t/512) * scale
// idx==null -> compacted frequencies k=m; useTwo -> s=2 except k==0 (irfft)
__global__ void k_build_basisT(const int* __restrict__ idx, float scale, int useTwo,
                               float* __restrict__ out, int total)
{
    int i = blockIdx.x * blockDim.x + threadIdx.x;
    if (i >= total) return;
    int t = i & 511;
    int m = i >> 9;
    int k = idx ? idx[m] : m;
    float s = (useTwo && k != 0) ? 2.f * scale : scale;
    int ph = (k * t) & 511;
    float ang = TWO_PI * (float)ph * (1.f / 512.f);
    out[m * 512 + t]        =  s * cosf(ang);
    out[(64 + m) * 512 + t] = -s * sinf(ang);
}

// per-mode complex mixing: O[b,h,o,m] = sum_e X[b,h,e,m] * W[h,e,o,m]
// X layout (transposed): (B, [re:0..63 | im:64..127] mode rows, 512 ch cols)
// O layout: (B, h*64+o, [re64|im64])  (A-operand of the inverse GEMM)
__global__ __launch_bounds__(256)
void k_modemix(const float* __restrict__ X, const float* __restrict__ wr,
               const float* __restrict__ wi, float* __restrict__ O, int total)
{
    int i = blockIdx.x * blockDim.x + threadIdx.x;
    if (i >= total) return;
    int m = i & 63;
    int o = (i >> 6) & 63;
    int h = (i >> 12) & 7;
    int b = i >> 15;
    const float* xr0 = X + (long long)b * 65536 + (long long)m * 512 + h * 64;
    const float* xi0 = xr0 + 64 * 512;
    const float* wrb = wr + (long long)h * 262144 + o * 64 + m;
    const float* wib = wi + (long long)h * 262144 + o * 64 + m;
    float ar = 0.f, ai = 0.f;
    for (int e = 0; e < 64; ++e) {
        float xr = xr0[e], xi = xi0[e];
        float wre = wrb[(long long)e * 4096], wie = wib[(long long)e * 4096];
        ar += xr * wre - xi * wie;
        ai += xr * wie + xi * wre;
    }
    float* ob = O + (long long)b * 65536 + (long long)(h * 64 + o) * 128;
    ob[m] = ar;
    ob[64 + m] = ai;
}

// a[b,h,x,y] = ctanh( sum_e qf[b,h,e,x] * kf[b,h,e,y] ); QF/KF transposed layout
__global__ __launch_bounds__(256)
void k_cross_qk(const float* __restrict__ QF, const float* __restrict__ KF,
                float* __restrict__ AT, int total)
{
    int i = blockIdx.x * blockDim.x + threadIdx.x;
    if (i >= total) return;
    int y = i & 63;
    int x = (i >> 6) & 63;
    int h = (i >> 12) & 7;
    int b = i >> 15;
    const float* qr0 = QF + (long long)b * 65536 + (long long)x * 512 + h * 64;
    const float* qi0 = qr0 + 64 * 512;
    const float* kr0 = KF + (long long)b * 65536 + (long long)y * 512 + h * 64;
    const float* ki0 = kr0 + 64 * 512;
    float re = 0.f, im = 0.f;
    for (int e = 0; e < 64; ++e) {
        float qr = qr0[e], qi = qi0[e];
        float kr = kr0[e], ki = ki0[e];
        re += qr * kr - qi * ki;
        im += qr * ki + qi * kr;
    }
    // complex tanh: tanh(a+ib) = [sinh 2a + i sin 2b] / [cosh 2a + cos 2b]
    float a2 = 2.f * re, b2 = 2.f * im;
    float ar, ai;
    if (fabsf(a2) > 30.f) { ar = a2 > 0.f ? 1.f : -1.f; ai = 0.f; }
    else {
        float den = coshf(a2) + cosf(b2);
        ar = sinhf(a2) / den;
        ai = sinf(b2) / den;
    }
    long long o = ((long long)(b * 8 + h) * 64 + x) * 64 + y;
    AT[o] = ar;
    AT[1048576 + o] = ai;
}

// v[b,h,e,x] = sum_y a[b,h,x,y] * kf[b,h,e,y]; output in transposed layout
__global__ __launch_bounds__(256)
void k_cross_av(const float* __restrict__ AT, const float* __restrict__ KF,
                float* __restrict__ VT, int total)
{
    int i = blockIdx.x * blockDim.x + threadIdx.x;
    if (i >= total) return;
    int x = i & 63;
    int e = (i >> 6) & 63;
    int h = (i >> 12) & 7;
    int b = i >> 15;
    const int c = h * 64 + e;
    const float* arb = AT + (long long)(b * 8 + h) * 4096 + (long long)x * 64;
    const float* aib = arb + 1048576;
    const float* kb  = KF + (long long)b * 65536 + c;
    float re = 0.f, im = 0.f;
    for (int y = 0; y < 64; ++y) {
        float ar = arb[y], ai = aib[y];
        float kr = kb[(long long)y * 512], ki = kb[(long long)(64 + y) * 512];
        re += ar * kr - ai * ki;
        im += ar * ki + ai * kr;
    }
    float* vb = VT + (long long)b * 65536 + c;
    vb[(long long)x * 512]        = re;
    vb[(long long)(64 + x) * 512] = im;
}

// final: out = trend[:, -256:] + my_ln(dec)[:, -256:] @ Wp + bp
__global__ __launch_bounds__(256)
void k_final(const float* __restrict__ LN, const float* __restrict__ Wp,
             const float* __restrict__ bp, const float* __restrict__ trend,
             float* __restrict__ out, int total)
{
    int i = blockIdx.x * blockDim.x + threadIdx.x;
    if (i >= total) return;
    int c = i % 21;
    int t = (i / 21) & 255;
    int b = i / (21 * 256);
    const float* xr = LN + ((long long)b * 512 + 256 + t) * 512;
    float acc = bp[c];
    for (int k = 0; k < 512; ++k) acc += xr[k] * Wp[k * 21 + c];
    out[i] = acc + trend[((long long)b * 512 + 256 + t) * 21 + c];
}

// =====================================================================
extern "C" void kernel_launch(void* const* d_in, const int* in_sizes, int n_in,
                              void* d_out, int out_size, void* d_ws, size_t ws_size,
                              hipStream_t stream)
{
    (void)in_sizes; (void)n_in; (void)out_size; (void)ws_size;
    auto F = [&](int i) { return (const float*)d_in[i]; };
    auto I = [&](int i) { return (const int*)d_in[i]; };

    const float* x_enc = F(0);
    const float* x_mark_enc = F(1);
    const float* x_mark_dec = F(3);          // x_dec (d_in[2]) unused by reference
    const float* enc_emb_conv = F(4);
    const float* enc_emb_time = F(5);
    const float* dec_emb_conv = F(6);
    const float* dec_emb_time = F(7);
    const float* eWq[2] = {F(8), F(18)};  const float* ebq[2] = {F(9), F(19)};
    const float* eWo[2] = {F(14), F(24)}; const float* ebo[2] = {F(15), F(25)};
    const float* eW1[2] = {F(16), F(26)}; const float* eW2[2] = {F(17), F(27)};
    const float* dWq = F(28), *dbq = F(29), *dWo = F(34), *dbo = F(35);
    const float* cWq = F(36), *cbq = F(37), *cWk = F(38), *cbk = F(39);
    const float* cWo = F(42), *cbo = F(43);
    const float* Wtrend = F(44), *dW1 = F(45), *dW2 = F(46);
    const float* enc_fr = F(47), *enc_fi = F(48);
    const float* ds_fr = F(49), *ds_fi = F(50), *dc_fr = F(51), *dc_fi = F(52);
    const float* eng = F(53), *enb = F(54), *dng = F(55), *dnb = F(56);
    const float* Wp = F(57), *bp = F(58);
    const int *idx_enc = I(59), *idx_dec = I(60), *idx_cq = I(61), *idx_ckv = I(62);

    // ---------------- workspace layout (floats) ----------------
    float* ws = (float*)d_ws;
    float* A0 = ws;                        // activation x        (32,512,512)
    float* A1 = A0 + 8388608;              // scratch / q / accum
    float* A2 = A1 + 8388608;              // scratch
    float* A3 = A2 + 8388608;              // encoder output
    float* A4 = A3 + 8388608;              // decoder trend accum (t1+t2+t3)
    float* XT = A4 + 8388608;              // spectral, transposed (B,128,512)
    float* OF = XT + 2097152;              // mixed spectrum (B,512,[re64|im64])
    float* XKT = OF + 2097152;             // cross keys, transposed
    float* AT = XKT + 2097152;             // cross attn map (re|im halves)
    float* VT = AT + 2097152;              // cross values, transposed
    float* TREND = VT + 2097152;           // (32,512,21)
    float* SEAS = TREND + 344064;
    float* TR0 = SEAS + 344064;            // trend0 then decoder seasonal input
    float* MARK = TR0 + 344064;            // (32,512,4)
    float* FW_ENC = MARK + 65536;          // forward bases, transposed (128,512)
    float* FW_DEC = FW_ENC + 65536;
    float* FW_CQ = FW_DEC + 65536;
    float* FW_CKV = FW_CQ + 65536;
    float* INV_SELF = FW_CKV + 65536;      // inverse bases (128,512)
    float* INV_CROSS = INV_SELF + 65536;

    const int TOT_BIG = 32 * 512 * 512;    // 8388608
    const int TOT_SMALL = 32 * 512 * 21;   // 344064
    const float invL = 1.f / 512.f;

    // ---------------- bases ----------------
    k_build_basisT<<<128, 256, 0, stream>>>(idx_enc, 1.f, 0, FW_ENC, 32768);
    k_build_basisT<<<128, 256, 0, stream>>>(idx_dec, 1.f, 0, FW_DEC, 32768);
    k_build_basisT<<<128, 256, 0, stream>>>(idx_cq, 1.f, 0, FW_CQ, 32768);
    k_build_basisT<<<128, 256, 0, stream>>>(idx_ckv, 1.f, 0, FW_CKV, 32768);
    k_build_basisT<<<128, 256, 0, stream>>>((const int*)nullptr, invL, 1, INV_SELF, 32768);
    k_build_basisT<<<128, 256, 0, stream>>>(idx_cq, invL, 1, INV_CROSS, 32768);

    // ---------------- input decomposition + trend init ----------------
    k_decomp<<<1344, 256, 0, stream>>>(x_enc, SEAS, TR0, 512, 21, 1, TOT_SMALL);
    k_trend_init<<<3, 256, 0, stream>>>(x_enc, TR0, TREND, 672);

    // ---------------- encoder embedding ----------------
    k_embed<<<32768, 256, 0, stream>>>(x_enc, enc_emb_conv, x_mark_enc, enc_emb_time,
                                       A0, 512, 21, 4, TOT_BIG);

    // helpers ----------------------------------------------------------
    auto self_fourier = [&](const float* Wq, const float* bq, const float* Wo,
                            const float* bo, const float* fr, const float* fi,
                            const float* FW) {
        // q = x @ Wq + bq  -> A1 (B,L,H*E)
        gemm(stream, A0, Wq, A1, bq, nullptr, 16384, 512, 512,
             512, 0, 512, 0, 512, 0, 1, GF_BIAS, 1.f);
        // transposed DFT at selected modes: XT[b] = FWT(128x512) @ q_b(512t x 512c)
        gemm(stream, FW, A1, XT, nullptr, nullptr, 128, 512, 512,
             512, 0, 512, 262144, 512, 65536, 32, 0, 1.f);
        k_modemix<<<4096, 256, 0, stream>>>(XT, fr, fi, OF, 1048576);
        // inverse (compacted modes k=0..63): y_b = OF_b(512c x 128) @ INV(128x512)
        gemm(stream, OF, INV_SELF, A2, nullptr, nullptr, 512, 512, 128,
             128, 65536, 512, 0, 512, 262144, 32, 0, 1.f);
        // FEDformer view-quirk: (B,H,E,L) reinterpreted flat as (B*L,512)
        gemm(stream, A2, Wo, A1, bo, A0, 16384, 512, 512,
             512, 0, 512, 0, 512, 0, 1, GF_BIAS | GF_RESID, 1.f);
    };
    auto ffn = [&](const float* W1, const float* W2) {
        hipMemcpyAsync(A1, A0, (size_t)TOT_BIG * sizeof(float),
                       hipMemcpyDeviceToDevice, stream);   // accum starts at residual x
        for (int c = 0; c < 4; ++c) {                      // chunk D_FF into 4x512
            gemm(stream, A0, W1 + c * 512, A2, nullptr, nullptr, 16384, 512, 512,
                 512, 0, 2048, 0, 512, 0, 1, GF_GELU, 1.f);
            gemm(stream, A2, W2 + (long long)c * 512 * 512, A1, nullptr, nullptr,
                 16384, 512, 512, 512, 0, 512, 0, 512, 0, 1, GF_ADDC, 1.f);
        }
    };

    // ---------------- encoder (2 layers) ----------------
    for (int l = 0; l < 2; ++l) {
        self_fourier(eWq[l], ebq[l], eWo[l], ebo[l], enc_fr, enc_fi, FW_ENC);
        k_decomp<<<32768, 256, 0, stream>>>(A1, A0, nullptr, 512, 512, 0, TOT_BIG);
        ffn(eW1[l], eW2[l]);
        k_decomp<<<32768, 256, 0, stream>>>(A1, A0, nullptr, 512, 512, 0, TOT_BIG);
    }
    k_layernorm<<<16384, 256, 0, stream>>>(A0, A1, eng, enb);
    k_sub_timemean<<<64, 256, 0, stream>>>(A1, A3, 16384);   // A3 = encoder output

    // ---------------- decoder embedding ----------------
    k_seasonal_init<<<1344, 256, 0, stream>>>(SEAS, TR0, TOT_SMALL);
    k_mark_concat<<<256, 256, 0, stream>>>(x_mark_enc, x_mark_dec, MARK, 65536);
    k_embed<<<32768, 256, 0, stream>>>(TR0, dec_emb_conv, MARK, dec_emb_time,
                                       A0, 512, 21, 4, TOT_BIG);

    // ---------------- decoder layer ----------------
    self_fourier(dWq, dbq, dWo, dbo, ds_fr, ds_fi, FW_DEC);
    k_decomp<<<32768, 256, 0, stream>>>(A1, A0, A4, 512, 512, 1, TOT_BIG);   // t1

    // cross attention (frequency domain)
    gemm(stream, A0, cWq, A1, cbq, nullptr, 16384, 512, 512,
         512, 0, 512, 0, 512, 0, 1, GF_BIAS, 1.f);
    gemm(stream, A3, cWk, A2, cbk, nullptr, 16384, 512, 512,
         512, 0, 512, 0, 512, 0, 1, GF_BIAS, 1.f);
    gemm(stream, FW_CQ, A1, XT, nullptr, nullptr, 128, 512, 512,
         512, 0, 512, 262144, 512, 65536, 32, 0, 1.f);
    gemm(stream, FW_CKV, A2, XKT, nullptr, nullptr, 128, 512, 512,
         512, 0, 512, 262144, 512, 65536, 32, 0, 1.f);
    k_cross_qk<<<4096, 256, 0, stream>>>(XT, XKT, AT, 1048576);
    k_cross_av<<<4096, 256, 0, stream>>>(AT, XKT, VT, 1048576);
    k_modemix<<<4096, 256, 0, stream>>>(VT, dc_fr, dc_fi, OF, 1048576);
    // inverse at idx_q frequencies; /(d_model^2) applied as f32 epilogue scale
    gemm(stream, OF, INV_CROSS, A2, nullptr, nullptr, 512, 512, 128,
         128, 65536, 512, 0, 512, 262144, 32, 0, 1.f / 262144.f);
    gemm(stream, A2, cWo, A1, cbo, A0, 16384, 512, 512,
         512, 0, 512, 0, 512, 0, 1, GF_BIAS | GF_RESID, 1.f);
    k_decomp<<<32768, 256, 0, stream>>>(A1, A0, A4, 512, 512, 2, TOT_BIG);   // t2

    ffn(dW1, dW2);
    k_decomp<<<32768, 256, 0, stream>>>(A1, A0, A4, 512, 512, 2, TOT_BIG);   // t3
    k_trend_conv<<<1344, 256, 0, stream>>>(A4, Wtrend, TREND, TOT_SMALL);

    // ---------------- head ----------------
    k_layernorm<<<16384, 256, 0, stream>>>(A0, A1, dng, dnb);
    k_sub_timemean<<<64, 256, 0, stream>>>(A1, A2, 16384);
    k_final<<<672, 256, 0, stream>>>(A2, Wp, bp, TREND, (float*)d_out, 172032);
}
